// EncoderDecoder_26731876450717
// MI455X (gfx1250) — matrix operand order repaired and checked by reading, not compiled, not run on hardware
//
#include <hip/hip_runtime.h>
#include <cstddef>

// ---------------------------------------------------------------------------
// GRU encoder-decoder forward (loss scalar), bf16 WMMA on gfx1250 (MI455X).
// B=256, S=64, T=20, E=100, H=1024, V=32000.
// ---------------------------------------------------------------------------

typedef __bf16 v16bf __attribute__((ext_vector_type(16)));
typedef __bf16 v8bf  __attribute__((ext_vector_type(8)));
typedef float  v8f   __attribute__((ext_vector_type(8)));

#define WMMA_BF16(a, b, c) \
    __builtin_amdgcn_wmma_f32_16x16x32_bf16(false, (a), false, (b), (short)0, (c), false, false)

// A-matrix 16x32 bf16 layout (contiguous, matching B): lanes 0-15 hold K={0..15},
// lanes 16-31 hold K={16..31}. Caller passes row pointer already offset
// by (lane>>4)*16 elements; we load 8 bf16 at +0 and 8 bf16 at +8.
__device__ __forceinline__ v16bf load_a16(const __bf16* p) {
    v8bf lo = *(const v8bf*)(p);
    v8bf hi = *(const v8bf*)(p + 8);
    return __builtin_shufflevector(lo, hi, 0,1,2,3,4,5,6,7,8,9,10,11,12,13,14,15);
}

__device__ __forceinline__ float sigmoidf_(float x) {
    return 1.0f / (1.0f + __expf(-x));
}

// ------------------------- utility kernels ---------------------------------

__global__ void init_out_kernel(float* o) { if (threadIdx.x == 0) o[0] = 0.0f; }

// dst[N][Kp] bf16 <- src[N][K] f32, zero-pad columns >= K
__global__ void cvt_pad_kernel(const float* __restrict__ src, __bf16* __restrict__ dst,
                               int N, int K, int Kp) {
    int idx = blockIdx.x * 256 + threadIdx.x;
    int total = N * Kp;
    if (idx >= total) return;
    int j = idx % Kp;
    int r = idx / Kp;
    float v = (j < K) ? src[(size_t)r * K + j] : 0.0f;
    dst[idx] = (__bf16)v;
}

// h0 f32 -> (f32 copy, bf16 copy)
__global__ void prep_h_kernel(const float* __restrict__ h0, float* __restrict__ hf,
                              __bf16* __restrict__ hb, int n) {
    int idx = blockIdx.x * 256 + threadIdx.x;
    if (idx >= n) return;
    float v = h0[idx];
    hf[idx] = v;
    hb[idx] = (__bf16)v;
}

// x_bf[19][256][1152] = { E_dec[token][0..99], ctx[b][0..1023], 0 x 28 } in bf16
__global__ void build_x_kernel(const float* __restrict__ Edec, const int* __restrict__ targets,
                               const float* __restrict__ ctx_f, __bf16* __restrict__ x) {
    int idx = blockIdx.x * 256 + threadIdx.x;
    const int total = 19 * 256 * 1152;
    if (idx >= total) return;
    int j = idx % 1152;
    int b = (idx / 1152) % 256;
    int t = idx / (1152 * 256);
    float v;
    if (j < 100) {
        int tok = (t == 0) ? 1 : targets[b * 20 + t];   // START_IDX = 1
        v = Edec[(size_t)tok * 100 + j];
    } else if (j < 1124) {
        v = ctx_f[b * 1024 + (j - 100)];
    } else {
        v = 0.0f;
    }
    x[idx] = (__bf16)v;
}

// ------------------------- encoder GRU step --------------------------------
// g = emb @ Wih^T + bih + h @ Whh^T + bhh ; gates ; h_new.
// Grid: 128 blocks x 256 thr (8 waves) -> 1024 wave-tiles = (M=256/16)x(H=1024/16).
__global__ __launch_bounds__(256)
void enc_step_kernel(const __bf16* __restrict__ hp_bf, const float* __restrict__ hp_f,
                     const __bf16* __restrict__ Eenc, const int* __restrict__ inputs,
                     const __bf16* __restrict__ Whh, const __bf16* __restrict__ Wih,
                     const float* __restrict__ bih, const float* __restrict__ bhh,
                     const int* __restrict__ lengths, int s,
                     float* __restrict__ hn_f, __bf16* __restrict__ hn_bf,
                     float* __restrict__ ctx_f, __bf16* __restrict__ ctx_bf) {
    constexpr int H = 1024;
    int lane = threadIdx.x & 31;
    int wave = threadIdx.x >> 5;
    int tile = blockIdx.x * 8 + wave;
    int m0 = (tile >> 6) << 4;
    int n0 = (tile & 63) << 4;
    int lr = lane & 15;
    int ao = (lane >> 4) << 4;   // A per-lane K offset (elements)
    int bo = (lane >> 4) << 4;   // B per-lane K offset (elements)

    v8f accR = {}, accZ = {}, accIN = {}, accHN = {};

    // hidden-side GEMM: K = 1024 (feeds r, z, hn)
    const __bf16* pa = hp_bf + (m0 + lr) * H + ao;
    const __bf16* pr = Whh + (size_t)(n0 + lr) * H + bo;
    const __bf16* pz = Whh + (size_t)(H + n0 + lr) * H + bo;
    const __bf16* pn = Whh + (size_t)(2 * H + n0 + lr) * H + bo;
    for (int k = 0; k < H; k += 32) {
        v16bf a  = load_a16(pa + k);
        v16bf br = *(const v16bf*)(pr + k);
        v16bf bz = *(const v16bf*)(pz + k);
        v16bf bn = *(const v16bf*)(pn + k);
        accR  = WMMA_BF16(a, br, accR);
        accZ  = WMMA_BF16(a, bz, accZ);
        accHN = WMMA_BF16(a, bn, accHN);
    }

    // input-side GEMM: K = 128 (padded E), gather bf16 embedding rows (feeds r, z, in)
    int tok = inputs[(m0 + lr) * 64 + s];
    const __bf16* pe = Eenc + (size_t)tok * 128 + ao;
    const __bf16* qr = Wih + (size_t)(n0 + lr) * 128 + bo;
    const __bf16* qz = Wih + (size_t)(H + n0 + lr) * 128 + bo;
    const __bf16* qn = Wih + (size_t)(2 * H + n0 + lr) * 128 + bo;
    for (int k = 0; k < 128; k += 32) {
        v16bf a  = load_a16(pe + k);
        v16bf br = *(const v16bf*)(qr + k);
        v16bf bz = *(const v16bf*)(qz + k);
        v16bf bn = *(const v16bf*)(qn + k);
        accR  = WMMA_BF16(a, br, accR);
        accZ  = WMMA_BF16(a, bz, accZ);
        accIN = WMMA_BF16(a, bn, accIN);
    }

    // epilogue: C layout -> row = m0 + v + (lane>=16 ? 8 : 0), col = n0 + (lane&15)
    int col = n0 + lr;
    float biasR = bih[col] + bhh[col];
    float biasZ = bih[H + col] + bhh[H + col];
    float bIN = bih[2 * H + col];
    float bHN = bhh[2 * H + col];
    int rbase = m0 + ((lane >> 4) << 3);
#pragma unroll
    for (int v = 0; v < 8; ++v) {
        int row = rbase + v;
        float r = sigmoidf_(accR[v] + biasR);
        float z = sigmoidf_(accZ[v] + biasZ);
        float n = tanhf(accIN[v] + bIN + r * (accHN[v] + bHN));
        float h = (1.0f - z) * n + z * hp_f[row * H + col];
        hn_f[row * H + col] = h;
        hn_bf[row * H + col] = (__bf16)h;
        if (lengths[row] == s + 1) {         // final hidden at step lengths-1
            ctx_f[row * H + col] = h;
            ctx_bf[row * H + col] = (__bf16)h;
        }
    }
}

// ------------------------- decoder GRU step --------------------------------
__global__ __launch_bounds__(256)
void dec_step_kernel(const __bf16* __restrict__ hp_bf, const float* __restrict__ hp_f,
                     const __bf16* __restrict__ x,  // [256][1152] bf16 for this step
                     const __bf16* __restrict__ Whh, const __bf16* __restrict__ Wih,
                     const float* __restrict__ bih, const float* __restrict__ bhh,
                     float* __restrict__ hn_f, __bf16* __restrict__ hn_bf) {
    constexpr int H = 1024, KX = 1152;
    int lane = threadIdx.x & 31;
    int wave = threadIdx.x >> 5;
    int tile = blockIdx.x * 8 + wave;
    int m0 = (tile >> 6) << 4;
    int n0 = (tile & 63) << 4;
    int lr = lane & 15;
    int ao = (lane >> 4) << 4;
    int bo = (lane >> 4) << 4;

    v8f accR = {}, accZ = {}, accIN = {}, accHN = {};

    const __bf16* pa = hp_bf + (m0 + lr) * H + ao;
    const __bf16* pr = Whh + (size_t)(n0 + lr) * H + bo;
    const __bf16* pz = Whh + (size_t)(H + n0 + lr) * H + bo;
    const __bf16* pn = Whh + (size_t)(2 * H + n0 + lr) * H + bo;
    for (int k = 0; k < H; k += 32) {
        v16bf a  = load_a16(pa + k);
        v16bf br = *(const v16bf*)(pr + k);
        v16bf bz = *(const v16bf*)(pz + k);
        v16bf bn = *(const v16bf*)(pn + k);
        accR  = WMMA_BF16(a, br, accR);
        accZ  = WMMA_BF16(a, bz, accZ);
        accHN = WMMA_BF16(a, bn, accHN);
    }

    const __bf16* px = x + (size_t)(m0 + lr) * KX + ao;
    const __bf16* qr = Wih + (size_t)(n0 + lr) * KX + bo;
    const __bf16* qz = Wih + (size_t)(H + n0 + lr) * KX + bo;
    const __bf16* qn = Wih + (size_t)(2 * H + n0 + lr) * KX + bo;
    for (int k = 0; k < KX; k += 32) {
        v16bf a  = load_a16(px + k);
        v16bf br = *(const v16bf*)(qr + k);
        v16bf bz = *(const v16bf*)(qz + k);
        v16bf bn = *(const v16bf*)(qn + k);
        accR  = WMMA_BF16(a, br, accR);
        accZ  = WMMA_BF16(a, bz, accZ);
        accIN = WMMA_BF16(a, bn, accIN);
    }

    int col = n0 + lr;
    float biasR = bih[col] + bhh[col];
    float biasZ = bih[H + col] + bhh[H + col];
    float bIN = bih[2 * H + col];
    float bHN = bhh[2 * H + col];
    int rbase = m0 + ((lane >> 4) << 3);
#pragma unroll
    for (int v = 0; v < 8; ++v) {
        int row = rbase + v;
        float r = sigmoidf_(accR[v] + biasR);
        float z = sigmoidf_(accZ[v] + biasZ);
        float n = tanhf(accIN[v] + bIN + r * (accHN[v] + bHN));
        float h = (1.0f - z) * n + z * hp_f[row * H + col];
        hn_f[row * H + col] = h;
        hn_bf[row * H + col] = (__bf16)h;
    }
}

// ------------------------- FC / logits -------------------------------------
// logits[256][32000] = h @ Wfc^T + bfc.
// Each wave computes a 16x64 output tile (4 accumulators sharing one A fetch):
// per K-chunk -> 1 A load (2xb128) + 4 B loads (8xb128) + 4 WMMAs.
// Tiles: 16 (M) x 500 (N/64) = 8000 wave-tiles -> 1000 blocks x 8 waves.
__global__ __launch_bounds__(256)
void fc_kernel(const __bf16* __restrict__ h_bf, const __bf16* __restrict__ Wfc,
               const float* __restrict__ bfc, float* __restrict__ logits) {
    constexpr int H = 1024, V = 32000;
    int lane = threadIdx.x & 31;
    int wave = threadIdx.x >> 5;
    int tile = blockIdx.x * 8 + wave;
    int mT = tile / 500, nT = tile % 500;
    int m0 = mT << 4, n0 = nT * 64;
    int lr = lane & 15;
    int ao = (lane >> 4) << 4;
    int bo = (lane >> 4) << 4;

    v8f acc0 = {}, acc1 = {}, acc2 = {}, acc3 = {};
    const __bf16* pa  = h_bf + (m0 + lr) * H + ao;
    const __bf16* pb0 = Wfc + (size_t)(n0 + lr) * H + bo;
    const __bf16* pb1 = Wfc + (size_t)(n0 + 16 + lr) * H + bo;
    const __bf16* pb2 = Wfc + (size_t)(n0 + 32 + lr) * H + bo;
    const __bf16* pb3 = Wfc + (size_t)(n0 + 48 + lr) * H + bo;
    for (int k = 0; k < H; k += 32) {
        v16bf a  = load_a16(pa + k);
        v16bf b0 = *(const v16bf*)(pb0 + k);
        v16bf b1 = *(const v16bf*)(pb1 + k);
        v16bf b2 = *(const v16bf*)(pb2 + k);
        v16bf b3 = *(const v16bf*)(pb3 + k);
        acc0 = WMMA_BF16(a, b0, acc0);
        acc1 = WMMA_BF16(a, b1, acc1);
        acc2 = WMMA_BF16(a, b2, acc2);
        acc3 = WMMA_BF16(a, b3, acc3);
    }
    int col = n0 + lr;
    float bias0 = bfc[col];
    float bias1 = bfc[col + 16];
    float bias2 = bfc[col + 32];
    float bias3 = bfc[col + 48];
    int rbase = m0 + ((lane >> 4) << 3);
#pragma unroll
    for (int v = 0; v < 8; ++v) {
        size_t rowoff = (size_t)(rbase + v) * V;
        logits[rowoff + col]      = acc0[v] + bias0;
        logits[rowoff + col + 16] = acc1[v] + bias1;
        logits[rowoff + col + 32] = acc2[v] + bias2;
        logits[rowoff + col + 48] = acc3[v] + bias3;
    }
}

// ------------------------- CE + loss ---------------------------------------
__global__ void ce_kernel(const float* __restrict__ logits, const int* __restrict__ targets,
                          int t, float* __restrict__ ce, float* __restrict__ mk) {
    __shared__ float red[256];
    int b = blockIdx.x;
    int tid = threadIdx.x;
    const float* row = logits + (size_t)b * 32000;

    float m = -3.402823e38f;
    for (int i = tid; i < 32000; i += 256) m = fmaxf(m, row[i]);
    red[tid] = m; __syncthreads();
    for (int s = 128; s > 0; s >>= 1) {
        if (tid < s) red[tid] = fmaxf(red[tid], red[tid + s]);
        __syncthreads();
    }
    m = red[0]; __syncthreads();

    float sum = 0.0f;
    for (int i = tid; i < 32000; i += 256) sum += __expf(row[i] - m);
    red[tid] = sum; __syncthreads();
    for (int s = 128; s > 0; s >>= 1) {
        if (tid < s) red[tid] += red[tid + s];
        __syncthreads();
    }
    if (tid == 0) {
        int y = targets[b * 20 + t + 1];
        float lse = m + __logf(red[0]);
        ce[b] = lse - row[y];
        mk[b] = (y >= 1) ? 1.0f : 0.0f;
    }
}

__global__ void loss_kernel(const float* __restrict__ ce, const float* __restrict__ mk,
                            float* __restrict__ out) {
    __shared__ float r1[256], r2[256];
    int tid = threadIdx.x;
    r1[tid] = ce[tid]; r2[tid] = mk[tid]; __syncthreads();
    for (int s = 128; s > 0; s >>= 1) {
        if (tid < s) { r1[tid] += r1[tid + s]; r2[tid] += r2[tid + s]; }
        __syncthreads();
    }
    if (tid == 0) out[0] += (r1[0] / 256.0f) * (r2[0] / 256.0f) / 20.0f;
}

// ---------------------------------------------------------------------------

extern "C" void kernel_launch(void* const* d_in, const int* in_sizes, int n_in,
                              void* d_out, int out_size, void* d_ws, size_t ws_size,
                              hipStream_t stream) {
    const int*   inputs  = (const int*)d_in[0];
    const int*   targets = (const int*)d_in[1];
    const int*   lengths = (const int*)d_in[2];
    const float* h0      = (const float*)d_in[3];
    const float* Eenc    = (const float*)d_in[4];
    const float* Edec    = (const float*)d_in[5];
    const float* Wihe    = (const float*)d_in[6];
    const float* Whhe    = (const float*)d_in[7];
    const float* bihe    = (const float*)d_in[8];
    const float* bhhe    = (const float*)d_in[9];
    const float* Wihd    = (const float*)d_in[10];
    const float* Whhd    = (const float*)d_in[11];
    const float* bihd    = (const float*)d_in[12];
    const float* bhhd    = (const float*)d_in[13];
    const float* Wfc     = (const float*)d_in[14];
    const float* bfc     = (const float*)d_in[15];
    float* out = (float*)d_out;

    char* ws = (char*)d_ws;
    size_t off = 0;
    auto alloc = [&](size_t bytes) -> void* {
        void* p = ws + off;
        off = (off + bytes + 255) & ~(size_t)255;
        return p;
    };
    __bf16* Wihe_b = (__bf16*)alloc((size_t)3072 * 128 * 2);
    __bf16* Whhe_b = (__bf16*)alloc((size_t)3072 * 1024 * 2);
    __bf16* Wihd_b = (__bf16*)alloc((size_t)3072 * 1152 * 2);
    __bf16* Whhd_b = (__bf16*)alloc((size_t)3072 * 1024 * 2);
    __bf16* Wfc_b  = (__bf16*)alloc((size_t)32000 * 1024 * 2);
    __bf16* Eenc_b = (__bf16*)alloc((size_t)32000 * 128 * 2);
    __bf16* x_b    = (__bf16*)alloc((size_t)19 * 256 * 1152 * 2);
    float*  hA_f   = (float*)alloc((size_t)256 * 1024 * 4);
    float*  hB_f   = (float*)alloc((size_t)256 * 1024 * 4);
    __bf16* hA_b   = (__bf16*)alloc((size_t)256 * 1024 * 2);
    __bf16* hB_b   = (__bf16*)alloc((size_t)256 * 1024 * 2);
    float*  ctx_f  = (float*)alloc((size_t)256 * 1024 * 4);
    __bf16* ctx_b  = (__bf16*)alloc((size_t)256 * 1024 * 2);
    float*  logits = (float*)alloc((size_t)256 * 32000 * 4);
    float*  ce     = (float*)alloc(256 * 4);
    float*  mk     = (float*)alloc(256 * 4);
    (void)ws_size; (void)n_in; (void)in_sizes; (void)out_size;

    init_out_kernel<<<1, 32, 0, stream>>>(out);

    auto cvt = [&](const float* src, __bf16* dst, int N, int K, int Kp) {
        int total = N * Kp;
        cvt_pad_kernel<<<(total + 255) / 256, 256, 0, stream>>>(src, dst, N, K, Kp);
    };
    cvt(Wihe, Wihe_b, 3072, 100, 128);
    cvt(Whhe, Whhe_b, 3072, 1024, 1024);
    cvt(Wihd, Wihd_b, 3072, 1124, 1152);
    cvt(Whhd, Whhd_b, 3072, 1024, 1024);
    cvt(Wfc,  Wfc_b,  32000, 1024, 1024);
    cvt(Eenc, Eenc_b, 32000, 100, 128);

    prep_h_kernel<<<(256 * 1024 + 255) / 256, 256, 0, stream>>>(h0, hA_f, hA_b, 256 * 1024);

    // ---- encoder: 64 sequential GRU steps (ping-pong h) ----
    float*  hf[2] = {hA_f, hB_f};
    __bf16* hb[2] = {hA_b, hB_b};
    int cur = 0;
    for (int s = 0; s < 64; ++s) {
        enc_step_kernel<<<128, 256, 0, stream>>>(
            hb[cur], hf[cur], Eenc_b, inputs, Whhe_b, Wihe_b, bihe, bhhe,
            lengths, s, hf[1 - cur], hb[1 - cur], ctx_f, ctx_b);
        cur ^= 1;
    }

    // ---- decoder inputs x = [demb, context] (bf16, padded) ----
    {
        int total = 19 * 256 * 1152;
        build_x_kernel<<<(total + 255) / 256, 256, 0, stream>>>(Edec, targets, ctx_f, x_b);
    }

    // ---- decoder: 19 steps of GRU + FC + CE + loss accumulation ----
    const float*  dpf = ctx_f;
    const __bf16* dpb = ctx_b;
    cur = 0;
    for (int t = 0; t < 19; ++t) {
        dec_step_kernel<<<128, 256, 0, stream>>>(
            dpb, dpf, x_b + (size_t)t * 256 * 1152, Whhd_b, Wihd_b, bihd, bhhd,
            hf[cur], hb[cur]);
        fc_kernel<<<1000, 256, 0, stream>>>(hb[cur], Wfc_b, bfc, logits);
        ce_kernel<<<256, 256, 0, stream>>>(logits, targets, t, ce, mk);
        loss_kernel<<<1, 256, 0, stream>>>(ce, mk, out);
        dpf = hf[cur];
        dpb = hb[cur];
        cur ^= 1;
    }
}
